// Attention_1451698946457
// MI455X (gfx1250) — matrix operand order repaired and checked by reading, not compiled, not run on hardware
//
#include <hip/hip_runtime.h>

// ---------------------------------------------------------------------------
// Problem constants (XLNet two-stream relative attention)
// ---------------------------------------------------------------------------
#define BB   2
#define SS   1024
#define S2   2048
#define PP   128
#define HH   16
#define DH   64
#define HID  1024
#define SCALE 0.125f      // 1/sqrt(64)
#define NEGINF (-1e9f)

typedef __attribute__((ext_vector_type(16))) __bf16 v16bf;
typedef __attribute__((ext_vector_type(8)))  float  v8f;
typedef __attribute__((ext_vector_type(4)))  unsigned int u32x4;

union TV { v16bf v; u32x4 q[2]; };

#define WMMA_BF16(A, B, C) \
  __builtin_amdgcn_wmma_f32_16x16x32_bf16(false, (A), false, (B), (short)0, (C), false, false)

__device__ inline v8f vzero8() {
  v8f z;
#pragma unroll
  for (int j = 0; j < 8; ++j) z[j] = 0.f;
  return z;
}

// A operand (16x32 M x K, bf16, row-major source, K contiguous).
// Layout (ISA 7.12.2): lanes 0-15 -> K 0..7 / 16..23; lanes 16-31 -> K 8..15 / 24..31.
__device__ inline v16bf ldA_bf(const __bf16* base, int stride, int lane) {
  int r = lane & 15, kh = (lane >> 4) & 1;
  const __bf16* p = base + (size_t)r * stride + kh * 8;
  TV t;
  t.q[0] = *(const u32x4*)(p);
  t.q[1] = *(const u32x4*)(p + 16);
  return t.v;
}

// B operand (32x16 K x N, bf16). Source is N x K row-major (row n, K contiguous).
// Layout (ISA 7.12.4 analog): lanes 0-15 -> K 0..7 / 16..23; lanes 16-31 -> K 8..15 / 24..31.
__device__ inline v16bf ldB_bf(const __bf16* base, int stride, int lane) {
  int n = lane & 15, kh = (lane >> 4) & 1;
  const __bf16* p = base + (size_t)n * stride + kh * 8;
  TV t;
  t.q[0] = *(const u32x4*)(p);
  t.q[1] = *(const u32x4*)(p + 16);
  return t.v;
}

// ---------------------------------------------------------------------------
// K-1a: elementwise f32 -> bf16 (activations)
// ---------------------------------------------------------------------------
__global__ void cvt_bf16(const float* __restrict__ X, __bf16* __restrict__ Y, int n) {
  int i = blockIdx.x * 256 + threadIdx.x;
  if (i < n) Y[i] = (__bf16)X[i];
}

// ---------------------------------------------------------------------------
// K-1b: LDS-tiled transpose+convert: W [K=1024][N=1024] f32 -> WT [N][K] bf16.
// Both read and write sides stay coalesced.
// ---------------------------------------------------------------------------
__global__ __launch_bounds__(256)
void transpose_cvt(const float* __restrict__ W, __bf16* __restrict__ WT) {
  __shared__ float tile[64][65];
  const int bx = blockIdx.x;            // n-tile
  const int by = blockIdx.y;            // k-tile
  const int tx = threadIdx.x & 63, ty = threadIdx.x >> 6;
  for (int r = ty; r < 64; r += 4)
    tile[r][tx] = W[(size_t)(by * 64 + r) * HID + bx * 64 + tx];
  __syncthreads();
  for (int r = ty; r < 64; r += 4)
    WT[(size_t)(bx * 64 + r) * HID + by * 64 + tx] = (__bf16)tile[tx][r];
}

// ---------------------------------------------------------------------------
// K0: recover one-hot target positions: qrow[b,p] = argmax_s target_mapping
// ---------------------------------------------------------------------------
__global__ void qrow_kernel(const float* __restrict__ tm, int* __restrict__ qrow) {
  int idx = blockIdx.x * blockDim.x + threadIdx.x;
  if (idx >= BB * PP) return;
  const float* row = tm + (size_t)idx * SS;
  int s = 0;
  for (int j = 0; j < SS; ++j)
    if (row[j] > 0.5f) s = j;
  qrow[idx] = s;
}

// ---------------------------------------------------------------------------
// K1: projection GEMM (all-bf16 operands, 16x64 tile per wave, A reuse x4)
// out[b,t,h,d](+bias) = sum_k X[b,t,k] * W[k, h*64+d]
// vt_mode=0: store [B][H][T][D]; vt_mode=1: store transposed [B][H][D][T].
// ---------------------------------------------------------------------------
__global__ __launch_bounds__(32)
void proj_gemm(const __bf16* __restrict__ Xb, const __bf16* __restrict__ WT, int T,
               const float* __restrict__ b0, const float* __restrict__ b1,
               const float* __restrict__ b2,
               __bf16* __restrict__ o0, __bf16* __restrict__ o1,
               __bf16* __restrict__ o2, int vt_mode) {
  const int lane = threadIdx.x;
  const int l15 = lane & 15, hi8 = (lane >> 4) * 8;
  const int m0 = blockIdx.x * 16;
  const int n0 = blockIdx.y * 64;

  v8f acc[4];
#pragma unroll
  for (int t = 0; t < 4; ++t) acc[t] = vzero8();

  const __bf16* Arow = Xb + (size_t)m0 * HID;
  for (int k0 = 0; k0 < HID; k0 += 32) {
    if (k0 + 32 < HID) {  // pull next k-step into cache (global_prefetch_b8)
      __builtin_prefetch(Arow + (size_t)l15 * HID + k0 + 32, 0, 1);
      __builtin_prefetch(WT + (size_t)(n0 + lane) * HID + k0 + 32, 0, 1);
      __builtin_prefetch(WT + (size_t)(n0 + 32 + lane) * HID + k0 + 32, 0, 1);
    }
    v16bf a = ldA_bf(Arow + k0, HID, lane);
#pragma unroll
    for (int nt = 0; nt < 4; ++nt) {
      v16bf bt = ldB_bf(WT + (size_t)(n0 + nt * 16) * HID + k0, HID, lane);
      acc[nt] = WMMA_BF16(a, bt, acc[nt]);
    }
  }

  const int bidx = m0 / T, tok0 = m0 - bidx * T;
#pragma unroll
  for (int nt = 0; nt < 4; ++nt) {
    const int n = n0 + nt * 16 + l15;
    const int h = n >> 6, d = n & 63;
    const float bb0 = b0 ? b0[n] : 0.f;
    const float bb1 = b1 ? b1[n] : 0.f;
    const float bb2 = b2 ? b2[n] : 0.f;
#pragma unroll
    for (int j = 0; j < 8; ++j) {
      int tok = tok0 + j + hi8;
      float v = acc[nt][j];
      size_t addr = vt_mode
          ? (((size_t)(bidx * HH + h) * DH + d) * T + tok)
          : (((size_t)(bidx * HH + h) * T + tok) * DH + d);
      o0[addr] = (__bf16)(v + bb0);
      if (o1) o1[addr] = (__bf16)(v + bb1);
      if (o2) o2[addr] = (__bf16)(v + bb2);
    }
  }
}

// ---------------------------------------------------------------------------
// K2: query-stream position matrix: ypos[b,h,p,c] = (qg+pb)[b,h,p,:] . kp[b,h,c,:]
// ---------------------------------------------------------------------------
__global__ __launch_bounds__(32)
void pos_query_gemm(const __bf16* __restrict__ qgpb, const __bf16* __restrict__ kp,
                    float* __restrict__ ypos) {
  const int lane = threadIdx.x;
  const int l15 = lane & 15, hi8 = (lane >> 4) * 8;
  int bid = blockIdx.x;
  const int pt = bid & 7; bid >>= 3;
  const int h = bid & 15;
  const int b = bid >> 4;
  const int p0 = pt * 16;
  const size_t bh = (size_t)(b * HH + h);

  const __bf16* A = qgpb + bh * (PP * DH) + (size_t)p0 * DH;
  const __bf16* kpb = kp + bh * (S2 * DH);
  float* yb = ypos + bh * ((size_t)PP * S2);

  v16bf a0 = ldA_bf(A, DH, lane);
  v16bf a1 = ldA_bf(A + 32, DH, lane);

  for (int ct = 0; ct < S2 / 16; ++ct) {
    if (ct + 1 < S2 / 16)
      __builtin_prefetch(kpb + (size_t)((ct + 1) * 16 + l15) * DH, 0, 1);
    v16bf b0 = ldB_bf(kpb + (size_t)(ct * 16) * DH, DH, lane);
    v16bf b1 = ldB_bf(kpb + (size_t)(ct * 16) * DH + 32, DH, lane);
    v8f d = vzero8();
    d = WMMA_BF16(a0, b0, d);
    d = WMMA_BF16(a1, b1, d);
#pragma unroll
    for (int j = 0; j < 8; ++j)
      yb[(size_t)(p0 + j + hi8) * S2 + ct * 16 + l15] = d[j];
  }
}

// ---------------------------------------------------------------------------
// K3: content-stream fused attention (one wave = one 16-row query tile).
// rel_shift via rolling aligned position-tile window; flash softmax in
// WMMA D-tile layout; PV through a 1KB LDS transpose bounce.
// ---------------------------------------------------------------------------
__global__ __launch_bounds__(32)
void attn_content(const __bf16* __restrict__ qcb, const __bf16* __restrict__ qpb,
                  const __bf16* __restrict__ qsb, const __bf16* __restrict__ kc,
                  const __bf16* __restrict__ kp, const __bf16* __restrict__ vT,
                  const float* __restrict__ segenc,
                  const unsigned char* __restrict__ segmat,
                  const float* __restrict__ mask, __bf16* __restrict__ attn_out) {
  const int lane = threadIdx.x;
  const int l15 = lane & 15;
  const int hi = lane >> 4;
  const int hi8 = hi * 8;

  int bid = blockIdx.x;
  const int qt = bid & 63; bid >>= 6;
  const int h = bid & 15;
  const int b = bid >> 4;
  const int i0 = qt * 16;
  const size_t bh = (size_t)(b * HH + h);

  const __bf16* qcb_bh = qcb + bh * (SS * DH) + (size_t)i0 * DH;
  const __bf16* qpb_bh = qpb + bh * (SS * DH) + (size_t)i0 * DH;
  const __bf16* qsb_bh = qsb + bh * (SS * DH) + (size_t)i0 * DH;
  const __bf16* kc_bh = kc + bh * (SS * DH);
  const __bf16* kp_bh = kp + bh * (S2 * DH);
  const __bf16* vT_bh = vT + bh * ((size_t)DH * SS);
  const unsigned char* sm_b = segmat + (size_t)b * SS * SS;
  const float* msk_b = mask + (size_t)b * SS * SS;

  __shared__ float posw[16][48];
  __shared__ __bf16 wbuf[16][32];
  __shared__ float segv[16][2];

  // per-row segment scores (seg bias already folded into qsb)
  {
    int r = l15, s = hi;
    const float* se = segenc + (size_t)(s * HH + h) * DH;
    float a = 0.f;
    for (int d = 0; d < DH; ++d) a += (float)qsb_bh[r * DH + d] * se[d];
    segv[r][s] = a;
  }
  __syncthreads();

  v16bf a_c0 = ldA_bf(qcb_bh, DH, lane);
  v16bf a_c1 = ldA_bf(qcb_bh + 32, DH, lane);
  v16bf a_p0 = ldA_bf(qpb_bh, DH, lane);
  v16bf a_p1 = ldA_bf(qpb_bh + 32, DH, lane);

  v8f acc[4];
#pragma unroll
  for (int t = 0; t < 4; ++t) acc[t] = vzero8();

  // initial previous position tile at columns [S - i0 - 16, S - i0)
  v8f p_prev;
  {
    int cprev = SS - i0 - 16;
    v16bf b0 = ldB_bf(kp_bh + (size_t)cprev * DH, DH, lane);
    v16bf b1 = ldB_bf(kp_bh + (size_t)cprev * DH + 32, DH, lane);
    p_prev = vzero8();
    p_prev = WMMA_BF16(a_p0, b0, p_prev);
    p_prev = WMMA_BF16(a_p1, b1, p_prev);
  }

  float mrun[8], lrun[8];
#pragma unroll
  for (int j = 0; j < 8; ++j) { mrun[j] = -1e30f; lrun[j] = 0.f; }

  for (int k0 = 0; k0 < SS; k0 += 32) {
    if (k0 + 32 < SS) {  // prefetch next key/value/pos rows
      __builtin_prefetch(kc_bh + (size_t)(k0 + 32 + lane) * DH, 0, 1);
      __builtin_prefetch(kp_bh + (size_t)(k0 + 32 + SS - i0 + lane) * DH, 0, 1);
      __builtin_prefetch(vT_bh + (size_t)(lane * 2) * SS + k0 + 32, 0, 1);
      __builtin_prefetch(vT_bh + (size_t)(lane * 2 + 1) * SS + k0 + 32, 0, 1);
    }

    // ---- content scores (2 x 16x16 D tiles over K=64)
    v16bf bk0a = ldB_bf(kc_bh + (size_t)k0 * DH, DH, lane);
    v16bf bk0b = ldB_bf(kc_bh + (size_t)k0 * DH + 32, DH, lane);
    v16bf bk1a = ldB_bf(kc_bh + (size_t)(k0 + 16) * DH, DH, lane);
    v16bf bk1b = ldB_bf(kc_bh + (size_t)(k0 + 16) * DH + 32, DH, lane);
    v8f c0 = vzero8(), c1 = vzero8();
    c0 = WMMA_BF16(a_c0, bk0a, c0);
    c0 = WMMA_BF16(a_c1, bk0b, c0);
    c1 = WMMA_BF16(a_c0, bk1a, c1);
    c1 = WMMA_BF16(a_c1, bk1b, c1);

    // ---- new position tiles at aligned columns cb, cb+16  (cb = k0 + S - i0)
    const int cb = k0 + SS - i0;
    v16bf bp0a = ldB_bf(kp_bh + (size_t)cb * DH, DH, lane);
    v16bf bp0b = ldB_bf(kp_bh + (size_t)cb * DH + 32, DH, lane);
    v16bf bp1a = ldB_bf(kp_bh + (size_t)(cb + 16) * DH, DH, lane);
    v16bf bp1b = ldB_bf(kp_bh + (size_t)(cb + 16) * DH + 32, DH, lane);
    v8f p0t = vzero8(), p1t = vzero8();
    p0t = WMMA_BF16(a_p0, bp0a, p0t);
    p0t = WMMA_BF16(a_p1, bp0b, p0t);
    p1t = WMMA_BF16(a_p0, bp1a, p1t);
    p1t = WMMA_BF16(a_p1, bp1b, p1t);

    // stage 48-column window [cb-16, cb+32) into LDS for diagonal gather
#pragma unroll
    for (int j = 0; j < 8; ++j) {
      int M = j + hi8;
      posw[M][l15] = p_prev[j];
      posw[M][16 + l15] = p0t[j];
      posw[M][32 + l15] = p1t[j];
    }
    __syncthreads();

    // ---- assemble scores:  pos[i,k] = window[r][16 + (k-k0) - r]
    float s0[8], s1[8];
#pragma unroll
    for (int j = 0; j < 8; ++j) {
      int M = j + hi8;
      int i = i0 + M;
      int kA = k0 + l15, kB = kA + 16;
      size_t rowoff = ((size_t)i << 10);
      float segA = sm_b[rowoff + kA] ? segv[M][1] : segv[M][0];
      float segB = sm_b[rowoff + kB] ? segv[M][1] : segv[M][0];
      float pA = posw[M][16 + l15 - M];
      float pB = posw[M][32 + l15 - M];
      s0[j] = (c0[j] + pA + segA) * SCALE + msk_b[rowoff + kA] * NEGINF;
      s1[j] = (c1[j] + pB + segB) * SCALE + msk_b[rowoff + kB] * NEGINF;
    }

    // ---- online softmax (row spread across 16 lanes of each half-wave)
    float rmax[8];
#pragma unroll
    for (int j = 0; j < 8; ++j) rmax[j] = fmaxf(s0[j], s1[j]);
    for (int mm = 1; mm < 16; mm <<= 1)
#pragma unroll
      for (int j = 0; j < 8; ++j)
        rmax[j] = fmaxf(rmax[j], __shfl_xor(rmax[j], mm, 32));

    float alpha[8], w0[8], w1[8], rsum[8];
#pragma unroll
    for (int j = 0; j < 8; ++j) {
      float mn = fmaxf(mrun[j], rmax[j]);
      alpha[j] = __expf(mrun[j] - mn);
      mrun[j] = mn;
      w0[j] = __expf(s0[j] - mn);
      w1[j] = __expf(s1[j] - mn);
      rsum[j] = w0[j] + w1[j];
    }
    for (int mm = 1; mm < 16; mm <<= 1)
#pragma unroll
      for (int j = 0; j < 8; ++j) rsum[j] += __shfl_xor(rsum[j], mm, 32);
#pragma unroll
    for (int j = 0; j < 8; ++j) lrun[j] = lrun[j] * alpha[j] + rsum[j];
#pragma unroll
    for (int t = 0; t < 4; ++t)
#pragma unroll
      for (int j = 0; j < 8; ++j) acc[t][j] *= alpha[j];

    // ---- D-layout weights -> LDS -> A-layout bf16 tile
#pragma unroll
    for (int j = 0; j < 8; ++j) {
      int M = j + hi8;
      wbuf[M][l15] = (__bf16)w0[j];
      wbuf[M][16 + l15] = (__bf16)w1[j];
    }
    __syncthreads();
    v16bf aw = ldA_bf(&wbuf[0][0], 32, lane);

    // ---- PV: acc[dt] += w(16x32) x V(32x16)
#pragma unroll
    for (int dt = 0; dt < 4; ++dt) {
      v16bf bv = ldB_bf(vT_bh + (size_t)(dt * 16) * SS + k0, SS, lane);
      acc[dt] = WMMA_BF16(aw, bv, acc[dt]);
    }
    p_prev = p1t;
    __syncthreads();
  }

  // ---- epilogue: normalize, store bf16 [B][S][H][D]
  float inv[8];
#pragma unroll
  for (int j = 0; j < 8; ++j) inv[j] = 1.0f / lrun[j];
#pragma unroll
  for (int dt = 0; dt < 4; ++dt)
#pragma unroll
    for (int j = 0; j < 8; ++j) {
      int i = i0 + j + hi8;
      attn_out[((size_t)(b * SS + i) * HH + h) * DH + dt * 16 + l15] =
          (__bf16)(acc[dt][j] * inv[j]);
    }
}

// ---------------------------------------------------------------------------
// K4: query-stream fused attention in compact P-space; position gathered from
// precomputed ypos (arbitrary per-row shift), rows mapped by qrow.
// ---------------------------------------------------------------------------
__global__ __launch_bounds__(32)
void attn_query(const __bf16* __restrict__ qgcb, const __bf16* __restrict__ qgsb,
                const __bf16* __restrict__ kc, const __bf16* __restrict__ vT,
                const float* __restrict__ ypos, const int* __restrict__ qrow,
                const float* __restrict__ segenc,
                const unsigned char* __restrict__ segmat,
                const float* __restrict__ mask, __bf16* __restrict__ attn_out) {
  const int lane = threadIdx.x;
  const int l15 = lane & 15;
  const int hi = lane >> 4;
  const int hi8 = hi * 8;

  int bid = blockIdx.x;
  const int pt = bid & 7; bid >>= 3;
  const int h = bid & 15;
  const int b = bid >> 4;
  const int p0 = pt * 16;
  const size_t bh = (size_t)(b * HH + h);

  const __bf16* A_c = qgcb + bh * (PP * DH) + (size_t)p0 * DH;
  const __bf16* A_s = qgsb + bh * (PP * DH) + (size_t)p0 * DH;
  const __bf16* kc_bh = kc + bh * (SS * DH);
  const __bf16* vT_bh = vT + bh * ((size_t)DH * SS);
  const float* yb = ypos + bh * ((size_t)PP * S2) + (size_t)p0 * S2;
  const unsigned char* sm_b = segmat + (size_t)b * SS * SS;
  const float* msk_b = mask + (size_t)b * SS * SS;

  __shared__ int irow[16];
  __shared__ __bf16 wbuf[16][32];
  __shared__ float segv[16][2];

  if (lane < 16) irow[lane] = qrow[b * PP + p0 + lane];
  __syncthreads();
  {
    int r = l15, s = hi;
    const float* se = segenc + (size_t)(s * HH + h) * DH;
    float a = 0.f;
    for (int d = 0; d < DH; ++d) a += (float)A_s[r * DH + d] * se[d];
    segv[r][s] = a;
  }
  __syncthreads();

  v16bf a_c0 = ldA_bf(A_c, DH, lane);
  v16bf a_c1 = ldA_bf(A_c + 32, DH, lane);

  v8f acc[4];
#pragma unroll
  for (int t = 0; t < 4; ++t) acc[t] = vzero8();
  float mrun[8], lrun[8];
#pragma unroll
  for (int j = 0; j < 8; ++j) { mrun[j] = -1e30f; lrun[j] = 0.f; }

  for (int k0 = 0; k0 < SS; k0 += 32) {
    if (k0 + 32 < SS) {
      __builtin_prefetch(kc_bh + (size_t)(k0 + 32 + lane) * DH, 0, 1);
      __builtin_prefetch(vT_bh + (size_t)(lane * 2) * SS + k0 + 32, 0, 1);
      __builtin_prefetch(vT_bh + (size_t)(lane * 2 + 1) * SS + k0 + 32, 0, 1);
    }
    v16bf bk0a = ldB_bf(kc_bh + (size_t)k0 * DH, DH, lane);
    v16bf bk0b = ldB_bf(kc_bh + (size_t)k0 * DH + 32, DH, lane);
    v16bf bk1a = ldB_bf(kc_bh + (size_t)(k0 + 16) * DH, DH, lane);
    v16bf bk1b = ldB_bf(kc_bh + (size_t)(k0 + 16) * DH + 32, DH, lane);
    v8f c0 = vzero8(), c1 = vzero8();
    c0 = WMMA_BF16(a_c0, bk0a, c0);
    c0 = WMMA_BF16(a_c1, bk0b, c0);
    c1 = WMMA_BF16(a_c0, bk1a, c1);
    c1 = WMMA_BF16(a_c1, bk1b, c1);

    float s0[8], s1[8];
#pragma unroll
    for (int j = 0; j < 8; ++j) {
      int M = j + hi8;
      int i = irow[M];
      int kA = k0 + l15, kB = kA + 16;
      size_t rowoff = ((size_t)i << 10);
      float segA = sm_b[rowoff + kA] ? segv[M][1] : segv[M][0];
      float segB = sm_b[rowoff + kB] ? segv[M][1] : segv[M][0];
      float pA = yb[(size_t)M * S2 + kA + SS - i];
      float pB = yb[(size_t)M * S2 + kB + SS - i];
      s0[j] = (c0[j] + pA + segA) * SCALE + msk_b[rowoff + kA] * NEGINF;
      s1[j] = (c1[j] + pB + segB) * SCALE + msk_b[rowoff + kB] * NEGINF;
    }

    float rmax[8];
#pragma unroll
    for (int j = 0; j < 8; ++j) rmax[j] = fmaxf(s0[j], s1[j]);
    for (int mm = 1; mm < 16; mm <<= 1)
#pragma unroll
      for (int j = 0; j < 8; ++j)
        rmax[j] = fmaxf(rmax[j], __shfl_xor(rmax[j], mm, 32));

    float alpha[8], w0[8], w1[8], rsum[8];
#pragma unroll
    for (int j = 0; j < 8; ++j) {
      float mn = fmaxf(mrun[j], rmax[j]);
      alpha[j] = __expf(mrun[j] - mn);
      mrun[j] = mn;
      w0[j] = __expf(s0[j] - mn);
      w1[j] = __expf(s1[j] - mn);
      rsum[j] = w0[j] + w1[j];
    }
    for (int mm = 1; mm < 16; mm <<= 1)
#pragma unroll
      for (int j = 0; j < 8; ++j) rsum[j] += __shfl_xor(rsum[j], mm, 32);
#pragma unroll
    for (int j = 0; j < 8; ++j) lrun[j] = lrun[j] * alpha[j] + rsum[j];
#pragma unroll
    for (int t = 0; t < 4; ++t)
#pragma unroll
      for (int j = 0; j < 8; ++j) acc[t][j] *= alpha[j];

#pragma unroll
    for (int j = 0; j < 8; ++j) {
      int M = j + hi8;
      wbuf[M][l15] = (__bf16)w0[j];
      wbuf[M][16 + l15] = (__bf16)w1[j];
    }
    __syncthreads();
    v16bf aw = ldA_bf(&wbuf[0][0], 32, lane);
#pragma unroll
    for (int dt = 0; dt < 4; ++dt) {
      v16bf bv = ldB_bf(vT_bh + (size_t)(dt * 16) * SS + k0, SS, lane);
      acc[dt] = WMMA_BF16(aw, bv, acc[dt]);
    }
    __syncthreads();
  }

  float inv[8];
#pragma unroll
  for (int j = 0; j < 8; ++j) inv[j] = 1.0f / lrun[j];
#pragma unroll
  for (int dt = 0; dt < 4; ++dt)
#pragma unroll
    for (int j = 0; j < 8; ++j) {
      int p = p0 + j + hi8;
      attn_out[((size_t)(b * PP + p) * HH + h) * DH + dt * 16 + l15] =
          (__bf16)(acc[dt][j] * inv[j]);
    }
}

// ---------------------------------------------------------------------------
// K5: output GEMM  O[m,n] = sum_k attn_bf16[m,k] * woT[n,k]   (16x64 tile/wave)
// ---------------------------------------------------------------------------
__global__ __launch_bounds__(32)
void out_gemm(const __bf16* __restrict__ A, const __bf16* __restrict__ WT,
              float* __restrict__ O) {
  const int lane = threadIdx.x;
  const int l15 = lane & 15, hi8 = (lane >> 4) * 8;
  const int m0 = blockIdx.x * 16;
  const int n0 = blockIdx.y * 64;

  v8f acc[4];
#pragma unroll
  for (int t = 0; t < 4; ++t) acc[t] = vzero8();

  const __bf16* Arow = A + (size_t)m0 * HID;
  for (int k0 = 0; k0 < HID; k0 += 32) {
    if (k0 + 32 < HID) {
      __builtin_prefetch(Arow + (size_t)l15 * HID + k0 + 32, 0, 1);
      __builtin_prefetch(WT + (size_t)(n0 + lane) * HID + k0 + 32, 0, 1);
      __builtin_prefetch(WT + (size_t)(n0 + 32 + lane) * HID + k0 + 32, 0, 1);
    }
    v16bf a = ldA_bf(Arow + k0, HID, lane);
#pragma unroll
    for (int nt = 0; nt < 4; ++nt) {
      v16bf bt = ldB_bf(WT + (size_t)(n0 + nt * 16) * HID + k0, HID, lane);
      acc[nt] = WMMA_BF16(a, bt, acc[nt]);
    }
  }
#pragma unroll
  for (int nt = 0; nt < 4; ++nt)
#pragma unroll
    for (int j = 0; j < 8; ++j)
      O[(size_t)(m0 + j + hi8) * HID + n0 + nt * 16 + l15] = acc[nt][j];
}

// ---------------------------------------------------------------------------
// Host launcher
// ---------------------------------------------------------------------------
extern "C" void kernel_launch(void* const* d_in, const int* in_sizes, int n_in,
                              void* d_out, int out_size, void* d_ws, size_t ws_size,
                              hipStream_t stream) {
  (void)in_sizes; (void)n_in; (void)out_size; (void)ws_size;

  const float* content = (const float*)d_in[0];
  const float* query   = (const float*)d_in[1];
  const float* posenc  = (const float*)d_in[2];
  const unsigned char* segmat = (const unsigned char*)d_in[3];  // bool
  const float* segenc  = (const float*)d_in[4];
  const float* segbias = (const float*)d_in[5];
  const float* cmask   = (const float*)d_in[6];
  const float* qmask   = (const float*)d_in[7];
  const float* tmap    = (const float*)d_in[8];
  const float* cbias   = (const float*)d_in[9];
  const float* pbias   = (const float*)d_in[10];
  const float* wq  = (const float*)d_in[11];
  const float* wkc = (const float*)d_in[12];
  const float* wv  = (const float*)d_in[13];
  const float* wkp = (const float*)d_in[14];
  const float* wo  = (const float*)d_in[15];
  float* out = (float*)d_out;

  size_t off = 0;
  char* ws = (char*)d_ws;
  auto take = [&](size_t bytes) -> void* {
    void* p = (void*)(ws + off);
    off = (off + bytes + 255) & ~(size_t)255;
    return p;
  };

  __bf16* qcb  = (__bf16*)take((size_t)BB * HH * SS * DH * 2);
  __bf16* qpb  = (__bf16*)take((size_t)BB * HH * SS * DH * 2);
  __bf16* qsb  = (__bf16*)take((size_t)BB * HH * SS * DH * 2);
  __bf16* kc   = (__bf16*)take((size_t)BB * HH * SS * DH * 2);
  __bf16* vT   = (__bf16*)take((size_t)BB * HH * DH * SS * 2);
  __bf16* kp   = (__bf16*)take((size_t)BB * HH * S2 * DH * 2);
  __bf16* qgcb = (__bf16*)take((size_t)BB * HH * PP * DH * 2);
  __bf16* qgpb = (__bf16*)take((size_t)BB * HH * PP * DH * 2);
  __bf16* qgsb = (__bf16*)take((size_t)BB * HH * PP * DH * 2);
  float*  ypos = (float*)take((size_t)BB * HH * PP * S2 * 4);
  __bf16* attn_c = (__bf16*)take((size_t)BB * SS * HH * DH * 2);
  __bf16* attn_q = (__bf16*)take((size_t)BB * PP * HH * DH * 2);
  int*    qrow = (int*)take((size_t)BB * PP * 4);
  // bf16 staging: activations + transposed weights
  __bf16* xc = (__bf16*)take((size_t)BB * SS * HID * 2);
  __bf16* xp = (__bf16*)take((size_t)BB * S2 * HID * 2);
  __bf16* xq = (__bf16*)take((size_t)BB * PP * HID * 2);
  __bf16* wqT  = (__bf16*)take((size_t)HID * HID * 2);
  __bf16* wkcT = (__bf16*)take((size_t)HID * HID * 2);
  __bf16* wvT  = (__bf16*)take((size_t)HID * HID * 2);
  __bf16* wkpT = (__bf16*)take((size_t)HID * HID * 2);
  __bf16* woT  = (__bf16*)take((size_t)HID * HID * 2);

  dim3 blk(32);
  dim3 tgrid(HID / 64, HID / 64);

  // one-time bf16 staging (coalesced)
  cvt_bf16<<<(BB * SS * HID + 255) / 256, 256, 0, stream>>>(content, xc, BB * SS * HID);
  cvt_bf16<<<(BB * S2 * HID + 255) / 256, 256, 0, stream>>>(posenc, xp, BB * S2 * HID);
  cvt_bf16<<<(BB * PP * HID + 255) / 256, 256, 0, stream>>>(query, xq, BB * PP * HID);
  transpose_cvt<<<tgrid, 256, 0, stream>>>(wq, wqT);
  transpose_cvt<<<tgrid, 256, 0, stream>>>(wkc, wkcT);
  transpose_cvt<<<tgrid, 256, 0, stream>>>(wv, wvT);
  transpose_cvt<<<tgrid, 256, 0, stream>>>(wkp, wkpT);
  transpose_cvt<<<tgrid, 256, 0, stream>>>(wo, woT);

  // target positions
  qrow_kernel<<<1, 256, 0, stream>>>(tmap, qrow);

  // projections (bf16 WMMA GEMMs with bias-fused epilogues)
  proj_gemm<<<dim3(BB * SS / 16, HID / 64), blk, 0, stream>>>(
      xc, wqT, SS, cbias, pbias, segbias, qcb, qpb, qsb, 0);
  proj_gemm<<<dim3(BB * SS / 16, HID / 64), blk, 0, stream>>>(
      xc, wkcT, SS, nullptr, nullptr, nullptr, kc, nullptr, nullptr, 0);
  proj_gemm<<<dim3(BB * SS / 16, HID / 64), blk, 0, stream>>>(
      xc, wvT, SS, nullptr, nullptr, nullptr, vT, nullptr, nullptr, 1);
  proj_gemm<<<dim3(BB * S2 / 16, HID / 64), blk, 0, stream>>>(
      xp, wkpT, S2, nullptr, nullptr, nullptr, kp, nullptr, nullptr, 0);
  proj_gemm<<<dim3(BB * PP / 16, HID / 64), blk, 0, stream>>>(
      xq, wqT, PP, cbias, pbias, segbias, qgcb, qgpb, qgsb, 0);

  // query-stream position matrix
  pos_query_gemm<<<BB * HH * (PP / 16), blk, 0, stream>>>(qgpb, kp, ypos);

  // fused attention (content + query streams)
  attn_content<<<BB * HH * (SS / 16), blk, 0, stream>>>(
      qcb, qpb, qsb, kc, kp, vT, segenc, segmat, cmask, attn_c);
  attn_query<<<BB * HH * (PP / 16), blk, 0, stream>>>(
      qgcb, qgsb, kc, vT, ypos, qrow, segenc, segmat, qmask, attn_q);

  // output projections
  out_gemm<<<dim3(BB * SS / 16, HID / 64), blk, 0, stream>>>(attn_c, woT, out);
  out_gemm<<<dim3(BB * PP / 16, HID / 64), blk, 0, stream>>>(
      attn_q, woT, out + (size_t)BB * SS * HID);
}
